// RegionProposalNetwork_3109556322619
// MI455X (gfx1250) — matrix operand order repaired and checked by reading, not compiled, not run on hardware
//
#include <hip/hip_runtime.h>

typedef __attribute__((ext_vector_type(8)))  float  v8f;
typedef __attribute__((ext_vector_type(16))) __bf16 v16bf;

union Frag { unsigned u[8]; uint4 q[2]; v16bf v; };

#define NIMG 2
#define CCH  256
#define CLIPV 4.135166556742356f

static __device__ __forceinline__ unsigned short f2bf(float f) {
  union { float f; unsigned u; } x; x.f = f;
  unsigned r = (x.u + 0x7FFFu + ((x.u >> 16) & 1u)) >> 16;
  return (unsigned short)r;
}

// ---------------------------------------------------------------------------
// Weight pre-pack: conv weights -> bf16 pairs in WMMA B-fragment layout.
// index = (((tap*8+cc)*16+cot)*32+lane)*8 + j ;  K pair = 2j + (lane>>4)*16
// ---------------------------------------------------------------------------
__global__ void packw_kernel(const float* __restrict__ Wc, unsigned* __restrict__ wpk) {
  int t = blockIdx.x * 256 + threadIdx.x;
  if (t >= 294912) return;
  int j = t & 7, lane = (t >> 3) & 31, cot = (t >> 8) & 15, cc = (t >> 12) & 7, tap = t >> 15;
  int k0 = 2 * j + (lane >> 4) * 16;
  int ci = cc * 32 + k0;
  int co = cot * 16 + (lane & 15);
  float w0 = Wc[(co * 256 + ci) * 9 + tap];
  float w1 = Wc[(co * 256 + ci + 1) * 9 + tap];
  wpk[t] = ((unsigned)f2bf(w1) << 16) | (unsigned)f2bf(w0);
}

// Head weights (3 logit rows + 12 reg rows, padded to 16) -> same B layout.
__global__ void packh_kernel(const float* __restrict__ Wl, const float* __restrict__ Wr,
                             unsigned* __restrict__ whpk) {
  int t = blockIdx.x * 256 + threadIdx.x;
  if (t >= 2048) return;
  int j = t & 7, lane = (t >> 3) & 31, cc = t >> 8;
  int k0 = 2 * j + (lane >> 4) * 16;
  int c = cc * 32 + k0;
  int oc = lane & 15;
  float w0 = 0.f, w1 = 0.f;
  if (oc < 3)       { w0 = Wl[oc * 256 + c];        w1 = Wl[oc * 256 + c + 1]; }
  else if (oc < 15) { w0 = Wr[(oc - 3) * 256 + c];  w1 = Wr[(oc - 3) * 256 + c + 1]; }
  whpk[t] = ((unsigned)f2bf(w1) << 16) | (unsigned)f2bf(w0);
}

// ---------------------------------------------------------------------------
// Fused 3x3 conv (implicit GEMM, WMMA bf16) + ReLU + logit/reg heads.
// Block: 256 thr (8 waves). Tile: 1 row, 64 positions, 256 out channels.
// 3-deep pipeline: async global->LDS f32 stage (N+2) || f32->bf16 convert (N+1)
// || WMMA compute (N).
// ---------------------------------------------------------------------------
#define ROWSZ (66 * 32)

static __device__ __forceinline__ void stage_async(
    const float* __restrict__ fmap, float* rowbuf, int n, int cc, int yy,
    int xb, int H, int W, int wave, int lane) {
  for (int ci = wave; ci < 32; ci += 8) {
    const float* rowg = fmap + ((n * CCH + cc * 32 + ci) * H + yy) * W;
    for (int col = lane; col < 66; col += 32) {
      int x = xb + col - 1;
      float* dst = &rowbuf[col * 32 + ci];
      if (yy >= 0 && yy < H && x >= 0 && x < W) {
        unsigned lds_off = (unsigned)(size_t)dst;
        unsigned long long ga = (unsigned long long)(size_t)(rowg + x);
        asm volatile("global_load_async_to_lds_b32 %0, %1, off"
                     :: "v"(lds_off), "v"(ga) : "memory");
      } else {
        *dst = 0.f;   // zero-pad (SAME conv halo / image edge)
      }
    }
  }
}

// Once-per-stage LDS f32 -> LDS bf16 convert (shared by all waves).
static __device__ __forceinline__ void convert_pass(
    const float* __restrict__ src, unsigned short* __restrict__ dst, int tid) {
  for (int e = tid * 2; e < ROWSZ; e += 512) {
    float f0 = src[e], f1 = src[e + 1];
    *(unsigned*)&dst[e] = ((unsigned)f2bf(f1) << 16) | (unsigned)f2bf(f0);
  }
}

__global__ __launch_bounds__(256) void conv_head_kernel(
    const float* __restrict__ fmap, const unsigned* __restrict__ wpk,
    const unsigned* __restrict__ whpk, const float* __restrict__ bconv,
    const float* __restrict__ blog, const float* __restrict__ breg,
    float* __restrict__ logits, float* __restrict__ regs, int H, int W) {
  __shared__ float rowf32[2][ROWSZ];          // [buf][col][ci] f32 async staging
  __shared__ unsigned short rowbf[2][ROWSZ];  // [buf][col][ci] bf16 for WMMA
  __shared__ unsigned short xtile[64 * 256];  // [pos][co] bf16 activations

  const int tid = threadIdx.x;
  const int lane = tid & 31, wave = tid >> 5;
  const int n = blockIdx.z, y = blockIdx.y, xb = blockIdx.x * 64;
  const int mb = (wave & 3) * 16;      // position tile base within 64
  const int cobase = (wave >> 2) * 8;  // first co-tile (of 16 ch) for this wave
  const int a_m = lane & 15;           // M row / N col within fragment
  const int a_hi = lane >> 4;          // lane half

  v8f acc[8];
#pragma unroll
  for (int t = 0; t < 8; t++) { v8f z = {}; acc[t] = z; }

  // Prologue: stage(0) -> wait -> convert(0); stage(1) in flight.
  stage_async(fmap, &rowf32[0][0], n, 0, y - 1, xb, H, W, wave, lane);
  asm volatile("s_wait_asynccnt 0x0" ::: "memory");
  __syncthreads();
  convert_pass(&rowf32[0][0], &rowbf[0][0], tid);
  stage_async(fmap, &rowf32[1][0], n, 0, y + 0, xb, H, W, wave, lane);  // it=1: cc=0,dy=1
  __syncthreads();

  for (int it = 0; it < 24; ++it) {          // it = cc*3 + dy
    int cc = it / 3, dy = it % 3;
    const unsigned short* rb = &rowbf[it & 1][0];
    for (int dx = 0; dx < 3; dx++) {         // kx taps reuse LDS via column shift
      Frag a;
#pragma unroll
      for (int j = 0; j < 8; j++) {          // A fragment: 2x ds_load_b128, no VALU
        int k0 = 2 * j + a_hi * 16;
        a.u[j] = *(const unsigned*)&rb[(mb + a_m + dx) * 32 + k0];
      }
      int tap = dy * 3 + dx;
      const uint4* pb = (const uint4*)wpk +
          ((((tap * 8 + cc) * 16 + cobase) * 32 + lane) << 1);
      Frag b[8];
#pragma unroll
      for (int t = 0; t < 8; t++) {          // 16 b128 loads, one clause
        b[t].q[0] = pb[t * 64];
        b[t].q[1] = pb[t * 64 + 1];
      }
      __builtin_amdgcn_sched_barrier(0);     // loads above, WMMA chain below
#pragma unroll
      for (int t = 0; t < 8; t++)            // back-to-back WMMA chain
        acc[t] = __builtin_amdgcn_wmma_f32_16x16x32_bf16(
            false, a.v, false, b[t].v, (short)0, acc[t], false, false);
    }
    if (it + 1 < 24) {
      asm volatile("s_wait_asynccnt 0x0" ::: "memory");  // stage(it+1) done
      __syncthreads();
      int q = (it + 1) & 1;
      convert_pass(&rowf32[q][0], &rowbf[q][0], tid);
      if (it + 2 < 24) {                     // stage(it+2) overlaps next compute
        int cc2 = (it + 2) / 3, dy2 = (it + 2) % 3;
        stage_async(fmap, &rowf32[it & 1][0], n, cc2, y + dy2 - 1, xb, H, W, wave, lane);
      }
      __syncthreads();
    }
  }

  // Epilogue: bias + ReLU -> bf16 activation tile in LDS.
  __syncthreads();
#pragma unroll
  for (int t = 0; t < 8; t++) {
    int co = (cobase + t) * 16 + a_m;
    float bb = bconv[co];
#pragma unroll
    for (int r = 0; r < 8; r++) {
      int pos = mb + r + a_hi * 8;
      float v = acc[t][r] + bb;
      xtile[pos * 256 + co] = f2bf(fmaxf(v, 0.f));
    }
  }
  __syncthreads();

  // Fused heads: out[pos, 15] = x[pos,:] @ Whead^T, one WMMA chain per M-tile.
  if (wave < 4) {
    int hmb = wave * 16;
    Frag hb[8];
    const uint4* ph = (const uint4*)whpk + (lane << 1);
#pragma unroll
    for (int c8 = 0; c8 < 8; c8++) {         // preload head B frags
      hb[c8].q[0] = ph[c8 * 64];
      hb[c8].q[1] = ph[c8 * 64 + 1];
    }
    v8f h = {};
#pragma unroll
    for (int c8 = 0; c8 < 8; c8++) {
      Frag a;
#pragma unroll
      for (int j = 0; j < 8; j++) {
        int k0 = 2 * j + a_hi * 16;
        a.u[j] = *(const unsigned*)&xtile[(hmb + a_m) * 256 + c8 * 32 + k0];
      }
      h = __builtin_amdgcn_wmma_f32_16x16x32_bf16(
          false, a.v, false, hb[c8].v, (short)0, h, false, false);
    }
    int hc = a_m;
#pragma unroll
    for (int r = 0; r < 8; r++) {
      int pos = hmb + r + a_hi * 8;
      int x = xb + pos;
      if (x < W) {
        int pidx = y * W + x;
        float v = h[r];
        if (hc < 3)
          logits[(n * H * W + pidx) * 3 + hc] = v + blog[hc];
        else if (hc < 15)
          regs[(n * H * W + pidx) * 12 + (hc - 3)] = v + breg[hc - 3];
      }
    }
  }
}

// ---------------------------------------------------------------------------
// Per-(image,level) top-500 via iterative block argmax (stable: lowest idx wins).
// Mutates the logit buffer (rewritten by conv each call -> deterministic).
// ---------------------------------------------------------------------------
__global__ __launch_bounds__(1024) void topk_kernel(float* __restrict__ logits,
                                                    float* __restrict__ topv,
                                                    int* __restrict__ topi) {
  const int LEN[5] = {182400, 45600, 11400, 2850, 741};
  const int OFF[5] = {0, 364800, 456000, 478800, 484500};
  int b = blockIdx.x, n = b / 5, lv = b % 5;
  float* L = logits + OFF[lv] + n * LEN[lv];
  int M = LEN[lv];
  __shared__ float sv[1024];
  __shared__ int si[1024];
  for (int k = 0; k < 500; k++) {
    float bv = -3.3e38f; int bi = 0x7fffffff;
    for (int i = threadIdx.x; i < M; i += 1024) {
      float v = L[i];
      if (v > bv || (v == bv && i < bi)) { bv = v; bi = i; }
    }
    sv[threadIdx.x] = bv; si[threadIdx.x] = bi;
    __syncthreads();
    for (int s = 512; s > 0; s >>= 1) {
      if (threadIdx.x < s) {
        float ov = sv[threadIdx.x + s]; int oi = si[threadIdx.x + s];
        if (ov > sv[threadIdx.x] || (ov == sv[threadIdx.x] && oi < si[threadIdx.x])) {
          sv[threadIdx.x] = ov; si[threadIdx.x] = oi;
        }
      }
      __syncthreads();
    }
    if (threadIdx.x == 0) {
      topv[b * 500 + k] = sv[0];
      topi[b * 500 + k] = si[0];
      L[si[0]] = -3.0e38f;
    }
    __syncthreads();
  }
}

// ---------------------------------------------------------------------------
// Decode boxes, sigmoid, clamp, validity; init sort keys (pad to 8192).
// ---------------------------------------------------------------------------
__global__ void decode_kernel(const float* __restrict__ topv, const int* __restrict__ topi,
                              const float* __restrict__ regs,
                              const float* pri0, const float* pri1, const float* pri2,
                              const float* pri3, const float* pri4,
                              const float* __restrict__ imsizes,
                              float* __restrict__ boxes, float* __restrict__ skey,
                              int* __restrict__ sidx, int* __restrict__ grp,
                              int* __restrict__ validf) {
  int e = blockIdx.x * 256 + threadIdx.x;
  if (e >= 8192) return;
  if (e >= 5000) { skey[e] = -1e30f; sidx[e] = e; return; }
  const int LEN[5] = {182400, 45600, 11400, 2850, 741};
  const int OFF[5] = {0, 364800, 456000, 478800, 484500};
  const float* pris[5] = {pri0, pri1, pri2, pri3, pri4};
  int n = e / 2500, p = e % 2500, lv = p / 500, kk = p % 500, b = n * 5 + lv;
  int idx = topi[b * 500 + kk];
  float lg = topv[b * 500 + kk];
  float score = 1.f / (1.f + expf(-lg));
  const float* pr = pris[lv] + idx * 4;
  const float* rg = regs + OFF[lv] * 4 + (n * LEN[lv] + idx) * 4;
  float pw = pr[2] - pr[0], ph = pr[3] - pr[1];
  float pcx = pr[0] + 0.5f * pw, pcy = pr[1] + 0.5f * ph;
  float dx = rg[0], dyv = rg[1];
  float dw = fminf(rg[2], CLIPV), dh = fminf(rg[3], CLIPV);
  float cx = dx * pw + pcx, cy = dyv * ph + pcy;
  float w = pw * expf(dw), h = ph * expf(dh);
  float x1 = cx - 0.5f * w, y1 = cy - 0.5f * h, x2 = cx + 0.5f * w, y2 = cy + 0.5f * h;
  float ih = imsizes[n * 2], iw = imsizes[n * 2 + 1];
  x1 = fminf(fmaxf(x1, 0.f), iw); x2 = fminf(fmaxf(x2, 0.f), iw);
  y1 = fminf(fmaxf(y1, 0.f), ih); y2 = fminf(fmaxf(y2, 0.f), ih);
  int valid = (score >= 0.05f) && ((x2 - x1) >= 1.f) && ((y2 - y1) >= 1.f);
  boxes[e * 4 + 0] = x1; boxes[e * 4 + 1] = y1;
  boxes[e * 4 + 2] = x2; boxes[e * 4 + 3] = y2;
  skey[e] = valid ? score : -1.f;
  sidx[e] = e;
  grp[e] = n * 10 + lv;
  validf[e] = valid;
}

// Bitonic sort, descending by key, ascending index tiebreak (stable argsort).
__global__ void bitonic_kernel(float* __restrict__ key, int* __restrict__ idx, int j, int k) {
  int i = blockIdx.x * 256 + threadIdx.x;
  int ixj = i ^ j;
  if (ixj <= i || i >= 8192) return;
  float ki = key[i], kj = key[ixj];
  int ii = idx[i], ij = idx[ixj];
  bool before = (ki > kj) || (ki == kj && ii < ij);
  bool desc = ((i & k) == 0);
  if (desc ? !before : before) {
    key[i] = kj; key[ixj] = ki; idx[i] = ij; idx[ixj] = ii;
  }
}

// ---------------------------------------------------------------------------
// Greedy NMS over sorted order + per-image top-300 + output write.
// ---------------------------------------------------------------------------
__global__ __launch_bounds__(512) void nms_final_kernel(
    const float* __restrict__ boxes, const int* __restrict__ grp,
    const int* __restrict__ validf, const int* __restrict__ sidx,
    float* __restrict__ out) {
  const int M = 5000;
  __shared__ unsigned char supp[5008];
  __shared__ signed char gsh[5008];
  __shared__ float bi[5];
  __shared__ int alive;
  int tid = threadIdx.x, nt = blockDim.x;
  for (int jj = tid; jj < M; jj += nt) {
    int p = sidx[jj];
    supp[jj] = validf[p] ? 0 : 1;
    gsh[jj] = (signed char)grp[p];
  }
  __syncthreads();
  for (int i = 0; i < M - 1; i++) {
    if (tid == 0) {
      int a = supp[i] ? 0 : 1;
      alive = a;
      if (a) {
        int p = sidx[i];
        bi[0] = boxes[p * 4]; bi[1] = boxes[p * 4 + 1];
        bi[2] = boxes[p * 4 + 2]; bi[3] = boxes[p * 4 + 3];
        bi[4] = (bi[2] - bi[0]) * (bi[3] - bi[1]);
      }
    }
    __syncthreads();
    if (alive) {
      float x1 = bi[0], y1 = bi[1], x2 = bi[2], y2 = bi[3], ai = bi[4];
      signed char gi = gsh[i];
      for (int jj = i + 1 + tid; jj < M; jj += nt) {
        if (supp[jj] || gsh[jj] != gi) continue;
        int p = sidx[jj];
        float bx1 = boxes[p * 4], by1 = boxes[p * 4 + 1];
        float bx2 = boxes[p * 4 + 2], by2 = boxes[p * 4 + 3];
        float lx = fmaxf(x1, bx1), ly = fmaxf(y1, by1);
        float rx = fminf(x2, bx2), ry = fminf(y2, by2);
        float iw = fmaxf(rx - lx, 0.f), ihh = fmaxf(ry - ly, 0.f);
        float inter = iw * ihh;
        float aj = (bx2 - bx1) * (by2 - by1);
        float iou = inter / (ai + aj - inter + 1e-9f);
        if (iou > 0.7f) supp[jj] = 1;
      }
    }
    __syncthreads();
  }
  if (tid == 0) {
    int cnt[2] = {0, 0};
    for (int jj = 0; jj < M; jj++) {
      int p = sidx[jj];
      int im = p / 2500;
      int kept = supp[jj] ? 0 : 1;
      int fin = 0;
      if (kept) { cnt[im]++; fin = (cnt[im] <= 300); }
      float f = fin ? 1.f : 0.f;
      out[jj * 4 + 0] = f * boxes[p * 4 + 0];
      out[jj * 4 + 1] = f * boxes[p * 4 + 1];
      out[jj * 4 + 2] = f * boxes[p * 4 + 2];
      out[jj * 4 + 3] = f * boxes[p * 4 + 3];
      out[20000 + jj] = fin ? (float)im : -1.f;
      out[25000 + jj] = f;
    }
  }
}

// ---------------------------------------------------------------------------
extern "C" void kernel_launch(void* const* d_in, const int* in_sizes, int n_in,
                              void* d_out, int out_size, void* d_ws, size_t ws_size,
                              hipStream_t stream) {
  (void)in_sizes; (void)n_in; (void)out_size; (void)ws_size;
  const float* fmaps[5]; for (int i = 0; i < 5; i++) fmaps[i] = (const float*)d_in[i];
  const float* pris[5];  for (int i = 0; i < 5; i++) pris[i] = (const float*)d_in[5 + i];
  const float* imsizes = (const float*)d_in[10];
  const float* Wc = (const float*)d_in[11];
  const float* bc = (const float*)d_in[12];
  const float* Wl = (const float*)d_in[13];
  const float* bl = (const float*)d_in[14];
  const float* Wr = (const float*)d_in[15];
  const float* br = (const float*)d_in[16];

  char* ws = (char*)d_ws;
  size_t off = 0;
  auto alloc = [&](size_t bytes) -> char* {
    char* p = ws + off; off += (bytes + 255) & ~(size_t)255; return p;
  };
  unsigned* wpk   = (unsigned*)alloc(294912u * 4);
  unsigned* whpk  = (unsigned*)alloc(2048u * 4);
  float*    logits= (float*)alloc(485982u * 4);
  float*    regs  = (float*)alloc(485982u * 16);
  float*    topv  = (float*)alloc(5000u * 4);
  int*      topi  = (int*)alloc(5000u * 4);
  float*    boxes = (float*)alloc(5000u * 16);
  float*    skey  = (float*)alloc(8192u * 4);
  int*      sidx  = (int*)alloc(8192u * 4);
  int*      grp   = (int*)alloc(5000u * 4);
  int*      validf= (int*)alloc(5000u * 4);

  packw_kernel<<<1152, 256, 0, stream>>>(Wc, wpk);
  packh_kernel<<<8, 256, 0, stream>>>(Wl, Wr, whpk);

  const int LH[5] = {200, 100, 50, 25, 13};
  const int LW[5] = {304, 152, 76, 38, 19};
  const int LVOFF[5] = {0, 364800, 456000, 478800, 484500};
  for (int lv = 0; lv < 5; lv++) {
    dim3 g((LW[lv] + 63) / 64, LH[lv], NIMG);
    conv_head_kernel<<<g, 256, 0, stream>>>(fmaps[lv], wpk, whpk, bc, bl, br,
                                            logits + LVOFF[lv],
                                            regs + (size_t)LVOFF[lv] * 4,
                                            LH[lv], LW[lv]);
  }
  topk_kernel<<<10, 1024, 0, stream>>>(logits, topv, topi);
  decode_kernel<<<32, 256, 0, stream>>>(topv, topi, regs, pris[0], pris[1], pris[2],
                                        pris[3], pris[4], imsizes, boxes, skey, sidx,
                                        grp, validf);
  for (int k = 2; k <= 8192; k <<= 1)
    for (int j = k >> 1; j > 0; j >>= 1)
      bitonic_kernel<<<32, 256, 0, stream>>>(skey, sidx, j, k);
  nms_final_kernel<<<1, 512, 0, stream>>>(boxes, grp, validf, sidx, (float*)d_out);
}
